// QuantizedLinear_53755810676756
// MI455X (gfx1250) — matrix operand order repaired and checked by reading, not compiled, not run on hardware
//
#include <hip/hip_runtime.h>

typedef __attribute__((ext_vector_type(16))) _Float16 v16h;
typedef __attribute__((ext_vector_type(2)))  _Float16 h2;
typedef __attribute__((ext_vector_type(8)))  float    v8f;
typedef __attribute__((ext_vector_type(4)))  float    f32x4;
typedef __attribute__((ext_vector_type(4)))  int      i32x4;
typedef __attribute__((ext_vector_type(8)))  unsigned u32x8;

#define IN_F   8192
#define OUT_F  8192
#define WAVES_PER_BLOCK 8
#define K_PER_ITER 32
#define K_BLOCK (WAVES_PER_BLOCK * K_PER_ITER)   /* 256 */
#define N_ITERS (IN_F / K_BLOCK)                 /* 32  */

// out[t, o] = scale[o] * sum_k x[t,k]*(w_q[o,k]-128) + bias[o]
// One block per 16-wide output-channel tile; 8 waves split K; WMMA f16->f32.
// Weight dequant uses the f16 magic-bias trick: for n in [0,1024),
// bits (0x6400 | n) == f16(1024 + n) exactly; subtracting 1152 (packed
// v_pk_add_f16) yields f16(n - 128) exactly. No v_cvt in the weight path.
__global__ __launch_bounds__(256) void qlinear_wmma_f16(
    const float* __restrict__ x,      // [16, 8192] f32
    const int*   __restrict__ wq,     // [8192, 8192] int32 codes 0..255
    const float* __restrict__ scale,  // [8192]
    const float* __restrict__ bias,   // [8192]
    float*       __restrict__ out)    // [16, 8192] f32
{
    // +1 pad to avoid LDS bank conflicts in the reduction read
    __shared__ float red[WAVES_PER_BLOCK][32][9];

    const int tid  = threadIdx.x;
    const int wave = tid >> 5;
    const int lane = tid & 31;
    const int n0   = blockIdx.x * 16;        // first output channel of tile
    const int mrow = lane & 15;              // A: row M / B: column N
    const int hi   = lane >> 4;              // lane-half selector

    const float* xrow = x + mrow * IN_F;                         // A source row
    const int*   wrow = wq + (size_t)(n0 + mrow) * (size_t)IN_F; // B column N

    const h2 c1152 = {(_Float16)1152.0f, (_Float16)1152.0f};

    v8f c = {};

    for (int it = 0; it < N_ITERS; ++it) {
        const int k0 = it * K_BLOCK + wave * K_PER_ITER;

        // ---- B fragment: 16 consecutive int32 weight codes ----
        // lanes 0-15: K = k0+0..15 of column N=lane; lanes 16-31: K = k0+16..31
        const int bbase = k0 + hi * 16;
        i32x4 w0 = *(const i32x4*)(wrow + bbase +  0);
        i32x4 w1 = *(const i32x4*)(wrow + bbase +  4);
        i32x4 w2 = *(const i32x4*)(wrow + bbase +  8);
        i32x4 w3 = *(const i32x4*)(wrow + bbase + 12);

        // prefetch next weight chunk for this wave (global_prefetch_b8)
        if (it + 1 < N_ITERS) {
            __builtin_prefetch((const void*)(wrow + bbase + K_BLOCK), 0, 0);
        }

        int wf[16];
        *(i32x4*)(wf +  0) = w0;
        *(i32x4*)(wf +  4) = w1;
        *(i32x4*)(wf +  8) = w2;
        *(i32x4*)(wf + 12) = w3;

        // magic-bias int->f16: (0x6400|n) is f16(1024+n); pk-sub 1152 -> n-128
        u32x8 bp;
#pragma unroll
        for (int i = 0; i < 8; ++i) {
            unsigned p = ((unsigned)wf[2 * i] |
                          ((unsigned)wf[2 * i + 1] << 16)) | 0x64006400u;
            h2 hv = __builtin_bit_cast(h2, p) - c1152;
            bp[i] = __builtin_bit_cast(unsigned, hv);
        }
        v16h b = __builtin_bit_cast(v16h, bp);

        // ---- A fragment: x row M, f32 -> f16 (v_cvt_pk_f16_f32) ----
        // lanes 0-15 hold K = k0 + 0..15; lanes 16-31: k0 + 16..31
        const int abase = k0 + hi * 16;
        f32x4 a0 = *(const f32x4*)(xrow + abase +  0);
        f32x4 a1 = *(const f32x4*)(xrow + abase +  4);
        f32x4 a2 = *(const f32x4*)(xrow + abase +  8);
        f32x4 a3 = *(const f32x4*)(xrow + abase + 12);
        float af[16];
        *(f32x4*)(af +  0) = a0;
        *(f32x4*)(af +  4) = a1;
        *(f32x4*)(af +  8) = a2;
        *(f32x4*)(af + 12) = a3;
        v16h a;
#pragma unroll
        for (int i = 0; i < 16; ++i) a[i] = (_Float16)af[i];

        // D(f32) = A(f16 16x32) x B(f16 32x16) + C
        c = __builtin_amdgcn_wmma_f32_16x16x32_f16(
                /*neg_a=*/false, a, /*neg_b=*/false, b,
                /*c_mod=*/(short)0, c, /*reuse_a=*/false, /*reuse_b=*/false);
    }

    // ---- cross-wave reduction of 8 partial 16x16 f32 tiles ----
#pragma unroll
    for (int v = 0; v < 8; ++v) red[wave][lane][v] = c[v];
    __syncthreads();

    // one thread per tile element: element (lane=l, vgpr=v) is [M = v + 8*(l>=16), N = l&15]
    const int v = tid >> 5;   // 0..7
    const int l = tid & 31;
    float acc = 0.0f;
#pragma unroll
    for (int w = 0; w < WAVES_PER_BLOCK; ++w) acc += red[w][l][v];

    const int M = v + ((l >> 4) << 3);
    const int o = n0 + (l & 15);
    out[M * OUT_F + o] = acc * scale[o] + bias[o];
}

extern "C" void kernel_launch(void* const* d_in, const int* in_sizes, int n_in,
                              void* d_out, int out_size, void* d_ws, size_t ws_size,
                              hipStream_t stream) {
    (void)in_sizes; (void)n_in; (void)out_size; (void)d_ws; (void)ws_size;
    const float* x     = (const float*)d_in[0];   // [1,16,8192] f32
    const int*   wq    = (const int*)  d_in[1];   // [8192,8192] int32
    const float* scale = (const float*)d_in[2];   // [8192] f32
    const float* bias  = (const float*)d_in[3];   // [8192] f32
    float*       out   = (float*)d_out;           // [1,16,8192] f32

    dim3 grid(OUT_F / 16);   // 512 tiles
    dim3 block(256);         // 8 wave32 waves
    qlinear_wmma_f16<<<grid, block, 0, stream>>>(x, wq, scale, bias, out);
}
